// GRAAttentionBlock_5360119185527
// MI455X (gfx1250) — matrix operand order repaired and checked by reading, not compiled, not run on hardware
//
#include <hip/hip_runtime.h>
#include <hip/hip_bf16.h>

// ---------------------------------------------------------------------------
// Swin-v2 window attention block for MI455X (gfx1250, wave32, WMMA).
// - All GEMMs: v_wmma_f32_16x16x32_bf16 (bf16 in, f32 accum)
// - Weights pre-transposed at conversion so B-tiles stage with b128 copies
// - Async global->LDS staging (ASYNCcnt) + LDS double buffering when the
//   toolchain exposes the gfx1250 async builtins (guarded by __has_builtin)
// ---------------------------------------------------------------------------

typedef __attribute__((ext_vector_type(16))) __bf16 v16bf;
typedef __attribute__((ext_vector_type(8)))  float  v8f;

#define NTOK   256      // tokens per window (16x16)
#define NHEAD  16
#define HDIM   32
#define CDIM   512
#define HIDDIM 1024
#define NWIN   256      // B * (H/ws) * (W/ws) = 16*4*4
#define TTOT   65536    // NWIN * NTOK
#define HIMG   64
#define WIMG   64

#ifndef __has_builtin
#define __has_builtin(x) 0
#endif

#if __has_builtin(__builtin_amdgcn_global_load_async_to_lds_b128)
#define HAS_ASYNC 1
#else
#define HAS_ASYNC 0
#endif

union Frag { v16bf v; uint4 u[2]; };

#if HAS_ASYNC
// builtin signature (from diagnostics): (int4 AS1*, int4 AS3*, imm, imm)
typedef int v4i_ __attribute__((vector_size(16)));
typedef __attribute__((address_space(1))) v4i_* gp128;
typedef __attribute__((address_space(3))) v4i_* lp128;
// per-lane 16B async copy global -> LDS (tracked by ASYNCcnt)
__device__ __forceinline__ void async_cp16(const void* g, void* l) {
  __builtin_amdgcn_global_load_async_to_lds_b128((gp128)g, (lp128)l, 0, 0);
}
__device__ __forceinline__ void wait_async() {
#if __has_builtin(__builtin_amdgcn_s_wait_asynccnt)
  __builtin_amdgcn_s_wait_asynccnt(0);
#else
  asm volatile("s_wait_asynccnt 0x0" ::: "memory");
#endif
}
#endif

// token (t, channel c) -> flat index into NCHW x
__device__ __forceinline__ int tok2pix(int t, int c) {
  int win = t >> 8, n = t & 255;
  int b = win >> 4, wy = (win >> 2) & 3, wx = win & 3;
  int iy = n >> 4, ix = n & 15;
  int hg = wy * 16 + iy, wg = wx * 16 + ix;
  return ((b * CDIM + c) * HIMG + hg) * WIMG + wg;
}

// ---------------------------------------------------------------------------
// 0) f32 [Kd][Nd] -> bf16 [Nd][Kd] transposed weight conversion (tiled)
// ---------------------------------------------------------------------------
__global__ __launch_bounds__(256) void cvtT_kernel(const float* __restrict__ src,
                                                   __bf16* __restrict__ dst,
                                                   int Kd, int Nd) {
  __shared__ float t[32][33];
  int n0 = blockIdx.x * 32, k0 = blockIdx.y * 32;
  int tx = threadIdx.x & 31, ty = threadIdx.x >> 5;  // 32 x 8
#pragma unroll
  for (int i = 0; i < 4; ++i)
    t[ty + 8 * i][tx] = src[(size_t)(k0 + ty + 8 * i) * Nd + n0 + tx];
  __syncthreads();
#pragma unroll
  for (int i = 0; i < 4; ++i)
    dst[(size_t)(n0 + ty + 8 * i) * Kd + k0 + tx] = (__bf16)t[tx][ty + 8 * i];
}

// ---------------------------------------------------------------------------
// 1) CPB MLP: tab[961][16] = relu(rel_table @ w1 + b1) @ w2
// ---------------------------------------------------------------------------
__global__ __launch_bounds__(256) void cpb_kernel(
    const float* __restrict__ rel_table, const float* __restrict__ w1,
    const float* __restrict__ b1, const float* __restrict__ w2,
    float* __restrict__ tab) {
  __shared__ float hid[512];
  __shared__ float red[256];
  int row = blockIdx.x, tid = threadIdx.x;
  float t0 = rel_table[row * 2 + 0], t1 = rel_table[row * 2 + 1];
  for (int j = tid; j < 512; j += 256)
    hid[j] = fmaxf(0.f, t0 * w1[j] + t1 * w1[512 + j] + b1[j]);
  __syncthreads();
  for (int h = 0; h < NHEAD; ++h) {
    red[tid] = hid[tid] * w2[tid * NHEAD + h] +
               hid[tid + 256] * w2[(tid + 256) * NHEAD + h];
    __syncthreads();
    for (int s = 128; s > 0; s >>= 1) {
      if (tid < s) red[tid] += red[tid + s];
      __syncthreads();
    }
    if (tid == 0) tab[row * NHEAD + h] = red[0];
    __syncthreads();
  }
}

// 2) expand: bias[h][q][k] = 16*sigmoid(tab[rel_index[q][k]][h])
__global__ void bias_kernel(const float* __restrict__ tab, const int* __restrict__ ridx,
                            float* __restrict__ bias) {
  int i = blockIdx.x * 256 + threadIdx.x;   // 16*256*256 total
  int h = i >> 16;
  int nm = i & 65535;
  int q = nm >> 8, k = nm & 255;
  float v = tab[ridx[q * 256 + k] * NHEAD + h];
  bias[i] = 16.f / (1.f + __expf(-v));
}

// ---------------------------------------------------------------------------
// 3) window partition NCHW -> token-major f32 residual stream
// ---------------------------------------------------------------------------
__global__ __launch_bounds__(256) void wpart_kernel(const float* __restrict__ x,
                                                    float* __restrict__ resid) {
  __shared__ float tile[16][17];
  int blk = blockIdx.x;                 // win*16 + iy
  int win = blk >> 4, iy = blk & 15;
  int b = win >> 4, wy = (win >> 2) & 3, wx = win & 3;
  int hg = wy * 16 + iy, wg0 = wx * 16;
  int tx = threadIdx.x & 15, ty = threadIdx.x >> 4;
  size_t tbase = (size_t)win * NTOK + iy * 16;
  for (int c0 = 0; c0 < CDIM; c0 += 16) {
    tile[ty][tx] = x[((size_t)(b * CDIM + c0 + ty) * HIMG + hg) * WIMG + wg0 + tx];
    __syncthreads();
    resid[(tbase + ty) * CDIM + c0 + tx] = tile[tx][ty];
    __syncthreads();
  }
}

// ---------------------------------------------------------------------------
// 4) LayerNorm over C=512, one block per token, output bf16
// ---------------------------------------------------------------------------
__global__ __launch_bounds__(256) void ln_kernel(const float* __restrict__ in,
                                                 const float* __restrict__ w,
                                                 const float* __restrict__ b,
                                                 __bf16* __restrict__ outp) {
  __shared__ float red[256];
  size_t t = blockIdx.x;
  int tid = threadIdx.x;
  float x0 = in[t * CDIM + tid];
  float x1 = in[t * CDIM + tid + 256];
  red[tid] = x0 + x1;
  __syncthreads();
  for (int s = 128; s > 0; s >>= 1) {
    if (tid < s) red[tid] += red[tid + s];
    __syncthreads();
  }
  float mean = red[0] * (1.f / 512.f);
  __syncthreads();
  float d0 = x0 - mean, d1 = x1 - mean;
  red[tid] = d0 * d0 + d1 * d1;
  __syncthreads();
  for (int s = 128; s > 0; s >>= 1) {
    if (tid < s) red[tid] += red[tid + s];
    __syncthreads();
  }
  float inv = rsqrtf(red[0] * (1.f / 512.f) + 1e-5f);
  outp[t * CDIM + tid]       = (__bf16)(d0 * inv * w[tid] + b[tid]);
  outp[t * CDIM + tid + 256] = (__bf16)(d1 * inv * w[tid + 256] + b[tid + 256]);
}

// ---------------------------------------------------------------------------
// 5) Generic WMMA GEMM: C[MxN] = A[MxK] @ B^T  (BT is pre-transposed [N][K]).
//    Block tile 128x64 (MODE 2: val+gate pair), BK=32, 8 waves.
//    MODE 0: QKV scatter (V written transposed for attention staging)
//    MODE 1: resid += A @ proj
//    MODE 2: mid = silu(gate) * val  (SwiGLU fused)
//    MODE 3: out[NCHW] = resid + A @ fc2 + x   (window reverse + outer skip)
// ---------------------------------------------------------------------------
template <int MODE>
__global__ __launch_bounds__(256) void gemm_kernel(
    const __bf16* __restrict__ A, const __bf16* __restrict__ BT,
    int K,
    __bf16* __restrict__ qd, __bf16* __restrict__ kd, __bf16* __restrict__ vd,
    float* __restrict__ resid, __bf16* __restrict__ mid,
    const float* __restrict__ xin, float* __restrict__ outp) {
  const int tid = threadIdx.x;
  const int lane = tid & 31, wv = tid >> 5;
  const int m0 = blockIdx.y * 128;
  const int j0 = blockIdx.x * 64;
  constexpr int nB = (MODE == 2) ? 2 : 1;

  v8f acc[2][4];
  const v8f vzero = {0.f, 0.f, 0.f, 0.f, 0.f, 0.f, 0.f, 0.f};
#pragma unroll
  for (int g = 0; g < 2; ++g)
#pragma unroll
    for (int jt = 0; jt < 4; ++jt) acc[g][jt] = vzero;

#if HAS_ASYNC
  // ---- async double-buffered k-loop -------------------------------------
  __shared__ __align__(16) __bf16 sA[2][128 * 48];
  __shared__ __align__(16) __bf16 sB[2][nB][64 * 48];
  auto stage = [&](int k0, int bi) {
    int row = tid >> 2, part = tid & 3;   // A: 128 rows x 4x8 elems, 2 passes
#pragma unroll
    for (int p = 0; p < 2; ++p) {
      int r = row + p * 64;
      async_cp16(A + (size_t)(m0 + r) * K + k0 + part * 8,
                 &sA[bi][r * 48 + part * 8]);
    }
#pragma unroll
    for (int g = 0; g < nB; ++g)          // B^T: 64 rows x 4x8 elems
      async_cp16(BT + (size_t)(j0 + g * HIDDIM + row) * K + k0 + part * 8,
                 &sB[bi][g][row * 48 + part * 8]);
  };
  stage(0, 0);
  int cur = 0;
  for (int k0 = 0; k0 < K; k0 += 32) {
    wait_async();          // own async copies into sX[cur] complete
    __syncthreads();       // ...and everyone else's; prior compute done too
    if (k0 + 32 < K) stage(k0 + 32, cur ^ 1);
    Frag a;
    {
      int m = wv * 16 + (lane & 15);
      int kh = (lane >> 4) * 16;
      a.u[0] = *(const uint4*)(&sA[cur][m * 48 + kh]);
      a.u[1] = *(const uint4*)(&sA[cur][m * 48 + kh + 8]);
    }
#pragma unroll
    for (int g = 0; g < nB; ++g)
#pragma unroll
      for (int jt = 0; jt < 4; ++jt) {
        Frag b;
        int nl = jt * 16 + (lane & 15);
        int kh = (lane >> 4) * 16;
        b.u[0] = *(const uint4*)(&sB[cur][g][nl * 48 + kh]);
        b.u[1] = *(const uint4*)(&sB[cur][g][nl * 48 + kh + 8]);
        acc[g][jt] = __builtin_amdgcn_wmma_f32_16x16x32_bf16(
            false, a.v, false, b.v, (short)0, acc[g][jt], false, false);
      }
    cur ^= 1;
  }
#else
  // ---- synchronous fallback ---------------------------------------------
  __shared__ __align__(16) __bf16 sA[128 * 48];
  __shared__ __align__(16) __bf16 sB[nB][64 * 48];
  for (int k0 = 0; k0 < K; k0 += 32) {
    {
      int row = tid >> 2, part = tid & 3;
#pragma unroll
      for (int p = 0; p < 2; ++p) {
        int r = row + p * 64;
        *(uint4*)(&sA[r * 48 + part * 8]) =
            *(const uint4*)(A + (size_t)(m0 + r) * K + k0 + part * 8);
      }
#pragma unroll
      for (int g = 0; g < nB; ++g)
        *(uint4*)(&sB[g][row * 48 + part * 8]) =
            *(const uint4*)(BT + (size_t)(j0 + g * HIDDIM + row) * K + k0 + part * 8);
    }
    __syncthreads();
    if (k0 + 32 < K)
      __builtin_prefetch(A + (size_t)(m0 + (tid >> 2)) * K + k0 + 32, 0, 0);
    Frag a;
    {
      int m = wv * 16 + (lane & 15);
      int kh = (lane >> 4) * 16;
      a.u[0] = *(const uint4*)(&sA[m * 48 + kh]);
      a.u[1] = *(const uint4*)(&sA[m * 48 + kh + 8]);
    }
#pragma unroll
    for (int g = 0; g < nB; ++g)
#pragma unroll
      for (int jt = 0; jt < 4; ++jt) {
        Frag b;
        int nl = jt * 16 + (lane & 15);
        int kh = (lane >> 4) * 16;
        b.u[0] = *(const uint4*)(&sB[g][nl * 48 + kh]);
        b.u[1] = *(const uint4*)(&sB[g][nl * 48 + kh + 8]);
        acc[g][jt] = __builtin_amdgcn_wmma_f32_16x16x32_bf16(
            false, a.v, false, b.v, (short)0, acc[g][jt], false, false);
      }
    __syncthreads();
  }
#endif

  // epilogue: D element r of lane l is (m = r + 8*(l>>4), n = l&15)
  const int hi = lane >> 4, nl = lane & 15;
  const int mbase = m0 + wv * 16 + 8 * hi;
#pragma unroll
  for (int jt = 0; jt < 4; ++jt) {
    int j = j0 + jt * 16 + nl;
#pragma unroll
    for (int r = 0; r < 8; ++r) {
      int t = mbase + r;
      float val = acc[0][jt][r];
      if (MODE == 0) {
        int which = j >> 9, head = (j >> 5) & 15, d = j & 31;
        int win = t >> 8, n = t & 255;
        if (which == 0) {
          qd[((size_t)(win * NHEAD + head) * NTOK + n) * HDIM + d] = (__bf16)val;
        } else if (which == 1) {
          kd[((size_t)(win * NHEAD + head) * NTOK + n) * HDIM + d] = (__bf16)val;
        } else {  // V stored transposed: [win*nh+head][d][n]
          vd[((size_t)(win * NHEAD + head) * HDIM + d) * NTOK + n] = (__bf16)val;
        }
      } else if (MODE == 1) {
        size_t idx = (size_t)t * CDIM + j;
        resid[idx] = resid[idx] + val;
      } else if (MODE == 2) {
        float g = acc[1][jt][r];
        float sil = g / (1.f + __expf(-g));  // silu(gate)
        mid[(size_t)t * HIDDIM + j] = (__bf16)(sil * val);
      } else {
        int xi = tok2pix(t, j);
        outp[xi] = resid[(size_t)t * CDIM + j] + val + xin[xi];
      }
    }
  }
}

// ---------------------------------------------------------------------------
// 6) Attention: one block per (window, head); 4 waves; flash-style chunks.
//    S = Q K^T * scale + bias; softmax; O = P V.  V arrives pre-transposed.
// ---------------------------------------------------------------------------
__global__ __launch_bounds__(128) void attn_kernel(
    const __bf16* __restrict__ q, const __bf16* __restrict__ kk,
    const __bf16* __restrict__ vT, const float* __restrict__ bias,
    __bf16* __restrict__ o) {
  __shared__ __align__(16) __bf16 sVT[32 * 264];      // V^T: [d][n]
  __shared__ __align__(16) __bf16 sP[4][16 * 264];    // per-wave P chunk
  const int tid = threadIdx.x, lane = tid & 31, wv = tid >> 5;
  const int wh = blockIdx.x;                 // win*NHEAD + head
  const int head = wh & 15, win = wh >> 4;
  const size_t base = (size_t)wh * NTOK * HDIM;
  const int hi = lane >> 4, nl = lane & 15;

  // stage V^T (already transposed in memory -> pure 16B copies)
#if HAS_ASYNC
#pragma unroll
  for (int p = 0; p < 8; ++p) {
    int id = p * 128 + tid;
    int d = id >> 5, seg = id & 31;
    async_cp16(vT + base + (size_t)d * NTOK + seg * 8, &sVT[d * 264 + seg * 8]);
  }
  wait_async();
#else
#pragma unroll
  for (int p = 0; p < 8; ++p) {
    int id = p * 128 + tid;
    int d = id >> 5, seg = id & 31;
    *(uint4*)(&sVT[d * 264 + seg * 8]) =
        *(const uint4*)(vT + base + (size_t)d * NTOK + seg * 8);
  }
#endif
  __syncthreads();

  const float scale = 0.17677669529663687f;  // 1/sqrt(32)
  for (int ch = wv; ch < 16; ch += 4) {      // uniform per wave
    int m0 = ch * 16;
    Frag a;   // Q fragment straight from global (rows are 64B)
    {
      int m = m0 + nl;
      int kh = hi * 16;
      a.u[0] = *(const uint4*)(q + base + (size_t)m * HDIM + kh);
      a.u[1] = *(const uint4*)(q + base + (size_t)m * HDIM + kh + 8);
    }
    v8f s[16];
#pragma unroll
    for (int jt = 0; jt < 16; ++jt) {
      Frag b;   // B = K^T == contiguous rows of K
      int n = jt * 16 + nl;
      int kh = hi * 16;
      b.u[0] = *(const uint4*)(kk + base + (size_t)n * HDIM + kh);
      b.u[1] = *(const uint4*)(kk + base + (size_t)n * HDIM + kh + 8);
      v8f z = {0.f, 0.f, 0.f, 0.f, 0.f, 0.f, 0.f, 0.f};
      s[jt] = __builtin_amdgcn_wmma_f32_16x16x32_bf16(
          false, a.v, false, b.v, (short)0, z, false, false);
    }
#pragma unroll
    for (int jt = 0; jt < 16; ++jt)
#pragma unroll
      for (int r = 0; r < 8; ++r) {
        int m = m0 + r + 8 * hi;
        s[jt][r] = s[jt][r] * scale +
                   bias[((size_t)head * NTOK + m) * NTOK + jt * 16 + nl];
      }
    float rinv[8];
#pragma unroll
    for (int r = 0; r < 8; ++r) {
      float mx = -1e30f;
#pragma unroll
      for (int jt = 0; jt < 16; ++jt) mx = fmaxf(mx, s[jt][r]);
#pragma unroll
      for (int d = 1; d < 16; d <<= 1) mx = fmaxf(mx, __shfl_xor(mx, d, 32));
      float sm = 0.f;
#pragma unroll
      for (int jt = 0; jt < 16; ++jt) {
        float e = __expf(s[jt][r] - mx);
        s[jt][r] = e;
        sm += e;
      }
#pragma unroll
      for (int d = 1; d < 16; d <<= 1) sm += __shfl_xor(sm, d, 32);
      rinv[r] = 1.f / sm;
    }
#pragma unroll
    for (int jt = 0; jt < 16; ++jt)
#pragma unroll
      for (int r = 0; r < 8; ++r) {
        int lr = r + 8 * hi;
        sP[wv][lr * 264 + jt * 16 + nl] = (__bf16)(s[jt][r] * rinv[r]);
      }
    v8f o0 = {0.f, 0.f, 0.f, 0.f, 0.f, 0.f, 0.f, 0.f};
    v8f o1 = {0.f, 0.f, 0.f, 0.f, 0.f, 0.f, 0.f, 0.f};
#pragma unroll
    for (int kb = 0; kb < 8; ++kb) {
      Frag pa;
      {
        int kh = hi * 16;
        pa.u[0] = *(const uint4*)(&sP[wv][nl * 264 + kb * 32 + kh]);
        pa.u[1] = *(const uint4*)(&sP[wv][nl * 264 + kb * 32 + kh + 8]);
      }
#pragma unroll
      for (int dt = 0; dt < 2; ++dt) {
        Frag vb;
        int dcol = dt * 16 + nl;
        int kh = hi * 16;
        vb.u[0] = *(const uint4*)(&sVT[dcol * 264 + kb * 32 + kh]);
        vb.u[1] = *(const uint4*)(&sVT[dcol * 264 + kb * 32 + kh + 8]);
        v8f oo = dt ? o1 : o0;
        oo = __builtin_amdgcn_wmma_f32_16x16x32_bf16(
            false, pa.v, false, vb.v, (short)0, oo, false, false);
        if (dt) o1 = oo; else o0 = oo;
      }
    }
#pragma unroll
    for (int dt = 0; dt < 2; ++dt) {
      v8f oo = dt ? o1 : o0;
#pragma unroll
      for (int r = 0; r < 8; ++r) {
        int m = m0 + r + 8 * hi;
        size_t t = (size_t)win * NTOK + m;
        o[t * CDIM + head * HDIM + dt * 16 + nl] = (__bf16)oo[r];
      }
    }
  }
}

// ---------------------------------------------------------------------------
// host: ws layout + launch sequence
// ---------------------------------------------------------------------------
extern "C" void kernel_launch(void* const* d_in, const int* in_sizes, int n_in,
                              void* d_out, int out_size, void* d_ws, size_t ws_size,
                              hipStream_t stream) {
  (void)in_sizes; (void)n_in; (void)out_size; (void)ws_size;
  const float* x       = (const float*)d_in[0];
  const float* rel_tab = (const float*)d_in[1];
  const float* n1w     = (const float*)d_in[2];
  const float* n1b     = (const float*)d_in[3];
  const float* n2w     = (const float*)d_in[4];
  const float* n2b     = (const float*)d_in[5];
  const float* qkv_w   = (const float*)d_in[6];
  const float* proj_w  = (const float*)d_in[7];
  const float* cpb_w1  = (const float*)d_in[8];
  const float* cpb_b1  = (const float*)d_in[9];
  const float* cpb_w2  = (const float*)d_in[10];
  const float* fc1_w   = (const float*)d_in[11];
  const float* fc2_w   = (const float*)d_in[12];
  const int*   ridx    = (const int*)d_in[13];
  float* out = (float*)d_out;

  char* p = (char*)d_ws;
  auto alloc = [&](size_t bytes) {
    char* r = p;
    p += (bytes + 255) & ~(size_t)255;
    return r;
  };
  float*  tab   = (float*)alloc((size_t)961 * NHEAD * 4);
  float*  bias  = (float*)alloc((size_t)NHEAD * NTOK * NTOK * 4);
  __bf16* wqkvT = (__bf16*)alloc((size_t)CDIM * 3 * CDIM * 2);   // [3C][C]
  __bf16* wprjT = (__bf16*)alloc((size_t)CDIM * CDIM * 2);       // [C][C]
  __bf16* wfc1T = (__bf16*)alloc((size_t)CDIM * 2 * HIDDIM * 2); // [2H][C]
  __bf16* wfc2T = (__bf16*)alloc((size_t)HIDDIM * CDIM * 2);     // [C][H]
  __bf16* hbuf  = (__bf16*)alloc((size_t)TTOT * CDIM * 2);
  __bf16* qbuf  = (__bf16*)alloc((size_t)TTOT * CDIM * 2);
  __bf16* kbuf  = (__bf16*)alloc((size_t)TTOT * CDIM * 2);
  __bf16* vbuf  = (__bf16*)alloc((size_t)TTOT * CDIM * 2);       // transposed
  __bf16* obuf  = (__bf16*)alloc((size_t)TTOT * CDIM * 2);
  float*  resid = (float*)alloc((size_t)TTOT * CDIM * 4);
  __bf16* mid   = (__bf16*)alloc((size_t)TTOT * HIDDIM * 2);

  // 0) weights -> bf16, transposed to [N][K]
  cvtT_kernel<<<dim3(3 * CDIM / 32, CDIM / 32), 256, 0, stream>>>(qkv_w, wqkvT, CDIM, 3 * CDIM);
  cvtT_kernel<<<dim3(CDIM / 32, CDIM / 32), 256, 0, stream>>>(proj_w, wprjT, CDIM, CDIM);
  cvtT_kernel<<<dim3(2 * HIDDIM / 32, CDIM / 32), 256, 0, stream>>>(fc1_w, wfc1T, CDIM, 2 * HIDDIM);
  cvtT_kernel<<<dim3(CDIM / 32, HIDDIM / 32), 256, 0, stream>>>(fc2_w, wfc2T, HIDDIM, CDIM);

  // 1-2) relative position bias
  cpb_kernel<<<961, 256, 0, stream>>>(rel_tab, cpb_w1, cpb_b1, cpb_w2, tab);
  bias_kernel<<<(NHEAD * NTOK * NTOK) / 256, 256, 0, stream>>>(tab, ridx, bias);

  // 3) window partition -> resid (token-major f32)
  wpart_kernel<<<NWIN * 16, 256, 0, stream>>>(x, resid);

  // 4) LN1 -> h (bf16)
  ln_kernel<<<TTOT, 256, 0, stream>>>(resid, n1w, n1b, hbuf);

  // 5) QKV projection (scatter; V transposed)
  gemm_kernel<0><<<dim3(3 * CDIM / 64, TTOT / 128), 256, 0, stream>>>(
      hbuf, wqkvT, CDIM, qbuf, kbuf, vbuf, nullptr, nullptr, nullptr, nullptr);

  // 6) attention
  attn_kernel<<<NWIN * NHEAD, 128, 0, stream>>>(qbuf, kbuf, vbuf, bias, obuf);

  // 7) output projection, resid += o @ proj
  gemm_kernel<1><<<dim3(CDIM / 64, TTOT / 128), 256, 0, stream>>>(
      obuf, wprjT, CDIM, nullptr, nullptr, nullptr, resid, nullptr, nullptr, nullptr);

  // 8) LN2 -> h (bf16, reuse buffer)
  ln_kernel<<<TTOT, 256, 0, stream>>>(resid, n2w, n2b, hbuf);

  // 9) fc1 + SwiGLU fused
  gemm_kernel<2><<<dim3(HIDDIM / 64, TTOT / 128), 256, 0, stream>>>(
      hbuf, wfc1T, CDIM, nullptr, nullptr, nullptr, nullptr, mid, nullptr, nullptr);

  // 10) fc2 + residual + window reverse + outer skip
  gemm_kernel<3><<<dim3(CDIM / 64, TTOT / 128), 256, 0, stream>>>(
      mid, wfc2T, HIDDIM, nullptr, nullptr, nullptr, resid, nullptr, x, out);
}
